// CAAN_84112639525655
// MI455X (gfx1250) — matrix operand order repaired and checked by reading, not compiled, not run on hardware
//
#include <hip/hip_runtime.h>

// ---------------------------------------------------------------------------
// CAAN (cross-asset attention) for MI455X / gfx1250, wave32 + WMMA bf16.
// N=4096, D_MODEL=512, DK=128, NUM_EMB=1024, QC=4.
//
// Flash-style fusion with a 128KB LDS score tile (CDNA5: 320KB LDS/WGP):
//   Pass A: gated scores, 4 waves split columns (computed exactly once)
//   Pass B: vectorized in-place exp(g - rowmax) (b128 LDS ops, 8 thr/row)
//   Pass C: p~ @ v via WMMA, A-frags served from LDS; normalize in epilogue
// ---------------------------------------------------------------------------

#define N_ASSETS 4096
#define DMODEL   512
#define DKDIM    128
#define NUMEMB   1024

// dynamic LDS layout (bytes)
#define SM_GATE  0
#define SM_S     4096                         // bf16[16*4096] = 131072
#define SM_MAX   (SM_S + 131072)              // float[4][16]
#define SM_SUMP  (SM_MAX + 256)               // float[128]
#define SM_PART  (SM_SUMP + 512)              // float[4][16]
#define SM_BYTES (SM_PART + 256)              // 136,192 B -> 2 blocks per WGP

typedef __attribute__((ext_vector_type(16))) __bf16 v16bf;
typedef __attribute__((ext_vector_type(8)))  float  v8f;

union ABfrag {          // 16 bf16 = 8 VGPRs, loaded as two 16B chunks
    v16bf v;
    uint4 u[2];
};

// ---------------------------------------------------------------------------
// gate_table[d] = sigmoid(rank_emb[d] . wL + wL_b)
// ---------------------------------------------------------------------------
__global__ void caan_gate_kernel(const float* __restrict__ emb,
                                 const float* __restrict__ wl,
                                 const float* __restrict__ wlb,
                                 float* __restrict__ gate) {
    int d = blockIdx.x * blockDim.x + threadIdx.x;
    if (d < NUMEMB) {
        float s = wlb[0];
        #pragma unroll 8
        for (int e = 0; e < 32; ++e) s += emb[d * 32 + e] * wl[e];
        gate[d] = 1.0f / (1.0f + __expf(-s));
    }
}

__global__ void caan_f32_to_bf16(const float* __restrict__ s,
                                 __bf16* __restrict__ d, int n) {
    int i = blockIdx.x * blockDim.x + threadIdx.x;
    if (i < n) d[i] = (__bf16)s[i];
}

// ---------------------------------------------------------------------------
// QKV projection: C = X @ W^T + b, all three weights fused along N.
// One wave per 16x16 tile. q,k stored row-major bf16; v stored transposed.
// ---------------------------------------------------------------------------
__global__ __launch_bounds__(32) void caan_qkv_kernel(
    const __bf16* __restrict__ Xb,
    const __bf16* __restrict__ Wqb, const __bf16* __restrict__ Wkb,
    const __bf16* __restrict__ Wvb,
    const float* __restrict__ bq, const float* __restrict__ bk,
    const float* __restrict__ bv,
    __bf16* __restrict__ qb, __bf16* __restrict__ kb,
    __bf16* __restrict__ vT) {
    const int lane = threadIdx.x & 31;
    const int half = lane >> 4;
    const int l16  = lane & 15;
    const int row0 = blockIdx.x * 16;
    const int n0   = blockIdx.y * 16;   // 0..767 over {q:128, k:128, v:512}

    const __bf16* W;
    const float*  bias;
    int nl;
    if (n0 < 128)      { W = Wqb; bias = bq; nl = n0; }
    else if (n0 < 256) { W = Wkb; bias = bk; nl = n0 - 128; }
    else               { W = Wvb; bias = bv; nl = n0 - 256; }

    v8f c = {0.f, 0.f, 0.f, 0.f, 0.f, 0.f, 0.f, 0.f};
    #pragma unroll 4
    for (int k0 = 0; k0 < DMODEL; k0 += 32) {
        ABfrag a, b;
        const __bf16* pa = Xb + (size_t)(row0 + l16) * DMODEL + k0 + half * 8;
        a.u[0] = *(const uint4*)pa;
        a.u[1] = *(const uint4*)(pa + 16);
        const __bf16* pb = W + (size_t)(nl + l16) * DMODEL + k0 + half * 8;
        b.u[0] = *(const uint4*)pb;
        b.u[1] = *(const uint4*)(pb + 16);
        c = __builtin_amdgcn_wmma_f32_16x16x32_bf16(false, a.v, false, b.v,
                                                    (short)0, c, false, false);
    }

    const float bcol = bias[nl + l16];
    #pragma unroll
    for (int i = 0; i < 8; ++i) {
        const int r = row0 + i + 8 * half;          // C layout: M = i + 8*half
        const __bf16 h = (__bf16)(c[i] + bcol);     // N = l16
        if (n0 < 128)      qb[(size_t)r * DKDIM + n0 + l16] = h;
        else if (n0 < 256) kb[(size_t)r * DKDIM + (n0 - 128) + l16] = h;
        else               vT[(size_t)(n0 - 256 + l16) * N_ASSETS + r] = h;
    }
}

// ---------------------------------------------------------------------------
// Fused gated flash attention + final projection.
// Block: 16 output rows, 4 waves. Columns split 4-ways for score generation;
// D_MODEL split 4-ways (128 each) for the p~ @ v accumulation.
// ---------------------------------------------------------------------------
__global__ __launch_bounds__(128) void caan_flash_kernel(
    const __bf16* __restrict__ qb, const __bf16* __restrict__ kb,
    const __bf16* __restrict__ vT, const float* __restrict__ ranks,
    const float* __restrict__ gate_tbl, const float* __restrict__ wf_w,
    const float* __restrict__ wf_b, float* __restrict__ out) {
    extern __shared__ __align__(16) char smem[];
    float*  sGate = (float*)(smem + SM_GATE);
    __bf16* sS    = (__bf16*)(smem + SM_S);     // [16][4096] scores -> probs
    float (*sMax)[16]  = (float(*)[16])(smem + SM_MAX);
    float*  sSumP      = (float*)(smem + SM_SUMP);       // [128] row partials
    float (*sPart)[16] = (float(*)[16])(smem + SM_PART);

    const int tid  = threadIdx.x;
    const int wave = tid >> 5;
    const int lane = tid & 31;
    const int half = lane >> 4;
    const int l16  = lane & 15;
    const int row0 = blockIdx.x * 16;
    const int d0   = wave * 128;       // this wave's D slice for pass C
    const int cq0  = wave * 1024;      // this wave's column quarter for pass A
    const float scale = 0.08838834764831845f;  // 1/sqrt(128)

    for (int i = tid; i < NUMEMB; i += 128) sGate[i] = gate_tbl[i];

    // Resident q A-frags for this row tile (K = 0..127).
    ABfrag qa[4];
    #pragma unroll
    for (int kk = 0; kk < 4; ++kk) {
        const __bf16* p = qb + (size_t)(row0 + l16) * DKDIM + kk * 32 + half * 8;
        qa[kk].u[0] = *(const uint4*)p;
        qa[kk].u[1] = *(const uint4*)(p + 16);
    }

    float rrow[8];
    #pragma unroll
    for (int i = 0; i < 8; ++i) rrow[i] = ranks[row0 + i + 8 * half];

    __syncthreads();   // sGate ready

    // gated 16x32 score tile (two 16x16 WMMA chains over K=128)
    auto gated_tile = [&](int c0, v8f& g0, v8f& g1) {
        #pragma unroll
        for (int t = 0; t < 2; ++t) {
            const int cb = c0 + 16 * t;
            v8f s = {0.f, 0.f, 0.f, 0.f, 0.f, 0.f, 0.f, 0.f};
            #pragma unroll
            for (int kk = 0; kk < 4; ++kk) {
                ABfrag b;
                const __bf16* p =
                    kb + (size_t)(cb + l16) * DKDIM + kk * 32 + half * 8;
                b.u[0] = *(const uint4*)p;
                b.u[1] = *(const uint4*)(p + 16);
                s = __builtin_amdgcn_wmma_f32_16x16x32_bf16(
                    false, qa[kk].v, false, b.v, (short)0, s, false, false);
            }
            const float rc = ranks[cb + l16];
            v8f g;
            #pragma unroll
            for (int i = 0; i < 8; ++i) {
                const float diff = fabsf(rrow[i] - rc);
                int di = (int)(diff * 0.25f);           // floor(diff/4), diff>=0
                di = di > (NUMEMB - 1) ? (NUMEMB - 1) : di;
                g[i] = sGate[di] * (s[i] * scale);
            }
            if (t == 0) g0 = g; else g1 = g;
        }
    };

    // ---- Pass A: gated scores for this wave's column quarter (once) ----
    float mlane[8];
    #pragma unroll
    for (int i = 0; i < 8; ++i) mlane[i] = -1e30f;

    for (int t0 = 0; t0 < 1024; t0 += 32) {
        const int c0 = cq0 + t0;
        v8f g0, g1;
        gated_tile(c0, g0, g1);
        #pragma unroll
        for (int i = 0; i < 8; ++i) {
            const int r = i + 8 * half;
            sS[r * N_ASSETS + c0 + l16]      = (__bf16)g0[i];
            sS[r * N_ASSETS + c0 + 16 + l16] = (__bf16)g1[i];
            mlane[i] = fmaxf(mlane[i], fmaxf(g0[i], g1[i]));
        }
    }
    #pragma unroll
    for (int i = 0; i < 8; ++i) {
        #pragma unroll
        for (int m = 8; m >= 1; m >>= 1)
            mlane[i] = fmaxf(mlane[i], __shfl_xor(mlane[i], m, 32));
    }
    if (l16 == 0) {
        #pragma unroll
        for (int i = 0; i < 8; ++i) sMax[wave][i + 8 * half] = mlane[i];
    }
    __syncthreads();   // scores + per-wave maxes visible

    // ---- Pass B: p~ = exp(g - rowmax) in place, vectorized b128 ----
    // 8 threads per row, 16B chunks; one scalar row-sum per thread.
    {
        const int r  = tid >> 3;          // row 0..15
        const int ch = tid & 7;           // chunk index within row
        const float m = fmaxf(fmaxf(sMax[0][r], sMax[1][r]),
                              fmaxf(sMax[2][r], sMax[3][r]));
        float ls = 0.f;
        __bf16* rowp = sS + r * N_ASSETS;
        for (int c = ch * 8; c < N_ASSETS; c += 64) {
            union { uint4 u; __bf16 h[8]; } f;
            f.u = *(uint4*)(rowp + c);
            #pragma unroll
            for (int e = 0; e < 8; ++e) {
                const float p = __expf((float)f.h[e] - m);
                ls += p;
                f.h[e] = (__bf16)p;
            }
            *(uint4*)(rowp + c) = f.u;
        }
        sSumP[tid] = ls;
    }
    __syncthreads();   // p~ tile + sum partials visible

    // ---- Pass C: acc += p~ @ v over all 4096 columns (A-frags from LDS) ----
    v8f acc[8];
    #pragma unroll
    for (int j = 0; j < 8; ++j)
        acc[j] = (v8f){0.f, 0.f, 0.f, 0.f, 0.f, 0.f, 0.f, 0.f};

    for (int c0 = 0; c0 < N_ASSETS; c0 += 32) {
        ABfrag pa;
        const __bf16* pr = sS + l16 * N_ASSETS + c0 + half * 8;   // A layout
        pa.u[0] = *(const uint4*)pr;
        pa.u[1] = *(const uint4*)(pr + 16);
        #pragma unroll
        for (int j = 0; j < 8; ++j) {
            ABfrag b;
            const __bf16* p =
                vT + (size_t)(d0 + 16 * j + l16) * N_ASSETS + c0 + half * 8;
            b.u[0] = *(const uint4*)p;
            b.u[1] = *(const uint4*)(p + 16);
            acc[j] = __builtin_amdgcn_wmma_f32_16x16x32_bf16(
                false, pa.v, false, b.v, (short)0, acc[j], false, false);
        }
    }

    // ---- Fused epilogue: out[r] = sigmoid((p~ @ v . wf) / l + b) ----
    float part[8];
    #pragma unroll
    for (int i = 0; i < 8; ++i) part[i] = 0.f;
    #pragma unroll
    for (int j = 0; j < 8; ++j) {
        const float w = wf_w[d0 + 16 * j + l16];
        #pragma unroll
        for (int i = 0; i < 8; ++i) part[i] += acc[j][i] * w;
    }
    #pragma unroll
    for (int i = 0; i < 8; ++i) {
        #pragma unroll
        for (int m = 8; m >= 1; m >>= 1)
            part[i] += __shfl_xor(part[i], m, 32);
    }
    if (l16 == 0) {
        #pragma unroll
        for (int i = 0; i < 8; ++i) sPart[wave][i + 8 * half] = part[i];
    }
    __syncthreads();
    if (tid < 16) {
        float l = 0.f;
        #pragma unroll
        for (int j = 0; j < 8; ++j) l += sSumP[tid * 8 + j];
        const float a = (sPart[0][tid] + sPart[1][tid] + sPart[2][tid] +
                         sPart[3][tid]) / l;
        out[row0 + tid] = 1.f / (1.f + __expf(-(a + wf_b[0])));
    }
}

// ---------------------------------------------------------------------------
extern "C" void kernel_launch(void* const* d_in, const int* in_sizes, int n_in,
                              void* d_out, int out_size, void* d_ws,
                              size_t ws_size, hipStream_t stream) {
    (void)in_sizes; (void)n_in; (void)out_size; (void)ws_size;
    const float* X     = (const float*)d_in[0];
    const float* ranks = (const float*)d_in[1];
    const float* Wq    = (const float*)d_in[2];
    const float* Wq_b  = (const float*)d_in[3];
    const float* Wk    = (const float*)d_in[4];
    const float* Wk_b  = (const float*)d_in[5];
    const float* Wv    = (const float*)d_in[6];
    const float* Wv_b  = (const float*)d_in[7];
    const float* emb   = (const float*)d_in[8];
    const float* wL    = (const float*)d_in[9];
    const float* wLb   = (const float*)d_in[10];
    const float* wf    = (const float*)d_in[11];
    const float* wfb   = (const float*)d_in[12];
    float* out = (float*)d_out;

    char* ws = (char*)d_ws;
    size_t off = 0;
    auto alloc = [&](size_t bytes) -> void* {
        void* p = ws + off;
        off = (off + bytes + 255) & ~(size_t)255;
        return p;
    };
    float*  gate = (float*)alloc((size_t)NUMEMB * 4);
    __bf16* Xb   = (__bf16*)alloc((size_t)N_ASSETS * DMODEL * 2);
    __bf16* Wqb  = (__bf16*)alloc((size_t)DKDIM * DMODEL * 2);
    __bf16* Wkb  = (__bf16*)alloc((size_t)DKDIM * DMODEL * 2);
    __bf16* Wvb  = (__bf16*)alloc((size_t)DMODEL * DMODEL * 2);
    __bf16* qbuf = (__bf16*)alloc((size_t)N_ASSETS * DKDIM * 2);
    __bf16* kbuf = (__bf16*)alloc((size_t)N_ASSETS * DKDIM * 2);
    __bf16* vT   = (__bf16*)alloc((size_t)DMODEL * N_ASSETS * 2);

    caan_gate_kernel<<<4, 256, 0, stream>>>(emb, wL, wLb, gate);

    int nX = N_ASSETS * DMODEL;
    caan_f32_to_bf16<<<(nX + 255) / 256, 256, 0, stream>>>(X, Xb, nX);
    int nQ = DKDIM * DMODEL;
    caan_f32_to_bf16<<<(nQ + 255) / 256, 256, 0, stream>>>(Wq, Wqb, nQ);
    caan_f32_to_bf16<<<(nQ + 255) / 256, 256, 0, stream>>>(Wk, Wkb, nQ);
    int nV = DMODEL * DMODEL;
    caan_f32_to_bf16<<<(nV + 255) / 256, 256, 0, stream>>>(Wv, Wvb, nV);

    // 768 fused output columns = q(128) | k(128) | v(512)
    caan_qkv_kernel<<<dim3(N_ASSETS / 16, 768 / 16), 32, 0, stream>>>(
        Xb, Wqb, Wkb, Wvb, Wq_b, Wk_b, Wv_b, qbuf, kbuf, vT);

    caan_flash_kernel<<<N_ASSETS / 16, 128, SM_BYTES, stream>>>(
        qbuf, kbuf, vT, ranks, gate, wf, wfb, out);
}
